// MultipathDynamicAttention_22754736735014
// MI455X (gfx1250) — matrix operand order repaired and checked by reading, not compiled, not run on hardware
//
#include <hip/hip_runtime.h>
#include <hip/hip_bf16.h>

// ---------------- WMMA types / helpers (gfx1250, wave32) ----------------
typedef __attribute__((ext_vector_type(16))) _Float16 v16h;
typedef __attribute__((ext_vector_type(8)))  _Float16 v8h;
typedef __attribute__((ext_vector_type(8)))  float    v8f;

#define BATCH 8
#define CCH   64
#define NTOK  4096
#define D3    192
#define KVAL  2048

union HPack { v16h v; v8h h[2]; };

// A-matrix 16x32 f16 loader (ISA 7.12.2): lane L holds row M=L%16;
// lo lanes: K={0..7,16..23}, hi lanes: K={8..15,24..31} -> two b128 loads.
__device__ __forceinline__ v16h load_A16x32(const _Float16* A, int lda, int lane) {
    int hs = lane >> 4, m = lane & 15;
    const _Float16* p = A + (size_t)m * lda + hs * 8;
    HPack u;
    u.h[0] = *(const v8h*)(p);
    u.h[1] = *(const v8h*)(p + 16);
    return u.v;
}

// B-matrix 32x16 f16 loader: lane L holds column N=L%16;
// lo lanes: K={0..7,16..23}, hi lanes: K={8..15,24..31}. Source must have K contiguous per column:
// element B(k,n) at Bc[n*ldb + k]  -> two b128 loads.
__device__ __forceinline__ v16h load_B32x16(const _Float16* Bc, int ldb, int lane) {
    int n = lane & 15, kh = (lane >> 4) * 8;
    const _Float16* p = Bc + (size_t)n * ldb + kh;
    HPack u;
    u.h[0] = *(const v8h*)(p);
    u.h[1] = *(const v8h*)(p + 16);
    return u.v;
}

__device__ __forceinline__ v8f wmma_f16(v16h a, v16h b, v8f c) {
    return __builtin_amdgcn_wmma_f32_16x16x32_f16(false, a, false, b,
                                                  (short)0, c, false, false);
}

// ---------------- 1) weight f32 -> f16 prepack ----------------
__global__ void cvt_f16_kernel(const float* __restrict__ s, _Float16* __restrict__ d, int n) {
    int i = blockIdx.x * 256 + threadIdx.x;
    if (i < n) d[i] = (_Float16)s[i];
}

// ---------------- 2) fused multi-path conv -> f_flat[b][n][192] (f16) ----
__global__ void conv_kernel(const float* __restrict__ x,
                            const float* __restrict__ w1, const float* __restrict__ b1,
                            const float* __restrict__ w3, const float* __restrict__ b3,
                            const float* __restrict__ w5, const float* __restrict__ b5,
                            _Float16* __restrict__ fh) {
    int t = blockIdx.x * 256 + threadIdx.x;       // t = ((b*64 + c)*4096 + n)
    int n = t & (NTOK - 1);
    int c = (t >> 12) & 63;
    int b = t >> 18;
    int yy = n >> 6, xx = n & 63;
    const float* xb = x + (size_t)b * CCH * NTOK;
    // 1x1 conv: x reads coalesced over n, w1 row broadcast per wave
    float f1 = b1[c];
    const float* w1c = w1 + c * 64;
#pragma unroll 8
    for (int ci = 0; ci < 64; ++ci) f1 += w1c[ci] * xb[(size_t)ci * NTOK + n];
    // depthwise 3x3 / 5x5, zero padding
    const float* xc = xb + (size_t)c * NTOK;
    float f2 = b3[c];
#pragma unroll
    for (int dy = -1; dy <= 1; ++dy)
#pragma unroll
        for (int dx = -1; dx <= 1; ++dx) {
            int Y = yy + dy, X = xx + dx;
            if ((unsigned)Y < 64u && (unsigned)X < 64u)
                f2 += w3[c * 9 + (dy + 1) * 3 + (dx + 1)] * xc[Y * 64 + X];
        }
    float f3 = b5[c];
#pragma unroll
    for (int dy = -2; dy <= 2; ++dy)
#pragma unroll
        for (int dx = -2; dx <= 2; ++dx) {
            int Y = yy + dy, X = xx + dx;
            if ((unsigned)Y < 64u && (unsigned)X < 64u)
                f3 += w5[c * 25 + (dy + 2) * 5 + (dx + 2)] * xc[Y * 64 + X];
        }
    _Float16* fo = fh + ((size_t)b * NTOK + n) * D3;
    fo[c]        = (_Float16)f1;
    fo[64 + c]   = (_Float16)f2;
    fo[128 + c]  = (_Float16)f3;
}

// ---------------- 3) QKV projection: [N,192] x [192,64] WMMA -----------
// 1/sqrt(E)=0.125 folded into Q (incl. bias). qw/kw/vw rows are already
// K-contiguous per output column -> direct B loads.
__global__ void qkv_kernel(const _Float16* __restrict__ fh,
                           const _Float16* __restrict__ qwh, const _Float16* __restrict__ kwh,
                           const _Float16* __restrict__ vwh,
                           const float* __restrict__ qb, const float* __restrict__ kb,
                           const float* __restrict__ vb,
                           _Float16* __restrict__ Qh, _Float16* __restrict__ Kh,
                           _Float16* __restrict__ Vh) {
    int gw = blockIdx.x * 4 + (threadIdx.x >> 5);
    int lane = threadIdx.x & 31;
    int mt  = gw & 255;
    int bat = (gw >> 8) & 7;
    int mat = gw >> 11;                        // 0:Q 1:K 2:V
    const _Float16* A = fh + ((size_t)bat * NTOK + mt * 16) * D3;
    const _Float16* W = (mat == 0) ? qwh : (mat == 1) ? kwh : vwh;
    const float* bias = (mat == 0) ? qb  : (mat == 1) ? kb  : vb;
    _Float16* dst = ((mat == 0) ? Qh : (mat == 1) ? Kh : Vh) + (size_t)bat * NTOK * 64;
    float scale = (mat == 0) ? 0.125f : 1.0f;
    v8f acc[4] = {};
    for (int kt = 0; kt < 6; ++kt) {
        v16h a = load_A16x32(A + kt * 32, D3, lane);
#pragma unroll
        for (int nt = 0; nt < 4; ++nt) {
            v16h b = load_B32x16(W + nt * 16 * D3 + kt * 32, D3, lane);
            acc[nt] = wmma_f16(a, b, acc[nt]);
        }
    }
    int nlo = lane & 15, mhi = (lane >> 4) * 8;
#pragma unroll
    for (int nt = 0; nt < 4; ++nt) {
        int e = nt * 16 + nlo;
        float bv = bias[e];
#pragma unroll
        for (int i = 0; i < 8; ++i) {
            int m = mt * 16 + mhi + i;
            dst[(size_t)m * 64 + e] = (_Float16)((acc[nt][i] + bv) * scale);
        }
    }
}

// ---------------- 4) attention logits: S = Q * K^T (f16, per batch) -----
// Each wave computes a 16x128 strip; A regs loaded once, reused 8x.
__global__ void logits_kernel(const _Float16* __restrict__ Qh, const _Float16* __restrict__ Kh,
                              _Float16* __restrict__ S, int bat) {
    int gw = blockIdx.x * 4 + (threadIdx.x >> 5);
    int lane = threadIdx.x & 31;
    int qt = gw >> 5;                                  // 0..255
    int kg = gw & 31;                                  // 0..31
    const _Float16* Qb = Qh + (size_t)bat * NTOK * 64 + (size_t)qt * 16 * 64;
    const _Float16* Kb = Kh + (size_t)bat * NTOK * 64;
    v16h a0 = load_A16x32(Qb + 0,  64, lane);
    v16h a1 = load_A16x32(Qb + 32, 64, lane);
    int nlo = lane & 15, mhi = (lane >> 4) * 8;
    for (int kt = 0; kt < 8; ++kt) {
        int key0 = kg * 128 + kt * 16;
        // B(k=e, n=key) = Kh[key][e] : rows of Kh are K-contiguous columns
        v16h b0 = load_B32x16(Kb + (size_t)key0 * 64 + 0,  64, lane);
        v16h b1 = load_B32x16(Kb + (size_t)key0 * 64 + 32, 64, lane);
        v8f acc = {};
        acc = wmma_f16(a0, b0, acc);
        acc = wmma_f16(a1, b1, acc);
#pragma unroll
        for (int i = 0; i < 8; ++i) {
            int q = qt * 16 + mhi + i;
            S[(size_t)q * NTOK + key0 + nlo] = (_Float16)acc[i];
        }
    }
}

// ---------------- 5) in-place row softmax (b128 vector accesses) --------
__global__ void softmax_kernel(_Float16* __restrict__ S) {
    __shared__ float red[256];
    int q = blockIdx.x, tid = threadIdx.x;
    _Float16* row = S + (size_t)q * NTOK;
    v8h c0 = *(const v8h*)(row + tid * 16);
    v8h c1 = *(const v8h*)(row + tid * 16 + 8);
    float v[16];
    float mx = -1e30f;
#pragma unroll
    for (int i = 0; i < 8; ++i) { v[i] = (float)c0[i]; v[8 + i] = (float)c1[i]; }
#pragma unroll
    for (int i = 0; i < 16; ++i) mx = fmaxf(mx, v[i]);
    red[tid] = mx; __syncthreads();
    for (int s = 128; s > 0; s >>= 1) { if (tid < s) red[tid] = fmaxf(red[tid], red[tid + s]); __syncthreads(); }
    mx = red[0]; __syncthreads();
    float sum = 0.f;
#pragma unroll
    for (int i = 0; i < 16; ++i) { v[i] = __expf(v[i] - mx); sum += v[i]; }
    red[tid] = sum; __syncthreads();
    for (int s = 128; s > 0; s >>= 1) { if (tid < s) red[tid] += red[tid + s]; __syncthreads(); }
    float inv = 1.0f / red[0];
#pragma unroll
    for (int i = 0; i < 8; ++i) { c0[i] = (_Float16)(v[i] * inv); c1[i] = (_Float16)(v[8 + i] * inv); }
    *(v8h*)(row + tid * 16)     = c0;
    *(v8h*)(row + tid * 16 + 8) = c1;
}

// ---------------- 6) key importance = column sums ----------------------
__global__ void colsum_kernel(const _Float16* __restrict__ S, float* __restrict__ colsum) {
    int k = blockIdx.x * 256 + threadIdx.x;
    float s = 0.f;
    for (int q = 0; q < NTOK; ++q) s += (float)S[(size_t)q * NTOK + k];
    colsum[k] = s;
}

// ---------------- 7) top-k (k=2048) via single-block bitonic sort -------
__global__ void topk_kernel(const float* __restrict__ colsum, float* __restrict__ mask) {
    __shared__ float arr[NTOK];                 // 16 KB of the 320 KB LDS
    int tid = threadIdx.x;                      // 1024 threads
    for (int i = tid; i < NTOK; i += 1024) arr[i] = colsum[i];
    __syncthreads();
    for (int k = 2; k <= NTOK; k <<= 1)
        for (int j = k >> 1; j > 0; j >>= 1) {
            for (int i = tid; i < NTOK; i += 1024) {
                int ixj = i ^ j;
                if (ixj > i) {
                    bool up = (i & k) == 0;     // descending sort
                    float a = arr[i], b = arr[ixj];
                    if (up ? (a < b) : (a > b)) { arr[i] = b; arr[ixj] = a; }
                }
            }
            __syncthreads();
        }
    float thresh = arr[KVAL - 1];
    for (int i = tid; i < NTOK; i += 1024)
        mask[i] = (colsum[i] >= thresh) ? 1.0f : 0.0f;
}

// ------- 8a) masked, transposed V: VmT[c][k] = mask[k]*V[k][c] ----------
__global__ void maskv_kernel(const _Float16* __restrict__ Vh, const float* __restrict__ mask,
                             _Float16* __restrict__ VmT, int bat) {
    int t = blockIdx.x * 256 + threadIdx.x;     // t = c*4096 + k
    int k = t & (NTOK - 1), c = t >> 12;
    VmT[t] = (_Float16)(mask[k] * (float)Vh[(size_t)bat * NTOK * 64 + (size_t)k * 64 + c]);
}

// ------- 8b) mask as a WMMA B-operand: maskB[16][4096], col 0 = mask ----
__global__ void maskb_kernel(const float* __restrict__ mask, _Float16* __restrict__ maskB) {
    int t = blockIdx.x * 256 + threadIdx.x;     // t = r*4096 + k, r in [0,16)
    int k = t & (NTOK - 1), r = t >> 12;
    maskB[t] = (r == 0) ? (_Float16)mask[k] : (_Float16)0.f;
}

// ---------------- 9) yh = (attn * Vm) / denom, denom fused via WMMA -----
// 5th accumulator tile with maskB as B computes denom[q] = sum_k a*mask in
// column 0; recovered with one cross-lane shuffle. Saves a full 33MB pass.
__global__ void av_kernel(const _Float16* __restrict__ S, const _Float16* __restrict__ VmT,
                          const _Float16* __restrict__ maskB, _Float16* __restrict__ yh) {
    int gw = blockIdx.x * 4 + (threadIdx.x >> 5);   // q-tile 0..255
    int lane = threadIdx.x & 31;
    const _Float16* A = S + (size_t)gw * 16 * NTOK;
    v8f acc[4] = {};
    v8f acc5 = {};
    for (int kt = 0; kt < 128; ++kt) {
        v16h a = load_A16x32(A + kt * 32, NTOK, lane);
#pragma unroll
        for (int nt = 0; nt < 4; ++nt) {
            v16h b = load_B32x16(VmT + (size_t)nt * 16 * NTOK + kt * 32, NTOK, lane);
            acc[nt] = wmma_f16(a, b, acc[nt]);
        }
        v16h bm = load_B32x16(maskB + kt * 32, NTOK, lane);
        acc5 = wmma_f16(a, bm, acc5);
    }
    int nlo = lane & 15, mhi = (lane >> 4) * 8;
#pragma unroll
    for (int i = 0; i < 8; ++i) {
        // denom for row m = mhi+i sits in column n=0: lane 0 (lo rows) / lane 16 (hi rows)
        float d = __shfl(acc5[i], lane & 16, 32) + 1e-8f;
        float inv = 1.0f / d;
        int q = gw * 16 + mhi + i;
#pragma unroll
        for (int nt = 0; nt < 4; ++nt)
            yh[(size_t)q * 64 + nt * 16 + nlo] = (_Float16)(acc[nt][i] * inv);
    }
}

// ---------------- 10) output projection + NCHW transpose store ----------
__global__ void proj_kernel(const _Float16* __restrict__ yh, const _Float16* __restrict__ owh,
                            const float* __restrict__ ob, float* __restrict__ out, int bat) {
    int gw = blockIdx.x * 4 + (threadIdx.x >> 5);   // q-tile 0..255
    int lane = threadIdx.x & 31;
    const _Float16* A = yh + (size_t)gw * 16 * 64;
    v8f acc[4] = {};
#pragma unroll
    for (int kt = 0; kt < 2; ++kt) {
        v16h a = load_A16x32(A + kt * 32, 64, lane);
#pragma unroll
        for (int nt = 0; nt < 4; ++nt) {
            v16h b = load_B32x16(owh + nt * 16 * 64 + kt * 32, 64, lane);
            acc[nt] = wmma_f16(a, b, acc[nt]);
        }
    }
    int nlo = lane & 15, mhi = (lane >> 4) * 8;
    float* ob_out = out + (size_t)bat * 64 * NTOK;
#pragma unroll
    for (int nt = 0; nt < 4; ++nt) {
        int o = nt * 16 + nlo;
        float bb = ob[o];
#pragma unroll
        for (int i = 0; i < 8; ++i) {
            int q = gw * 16 + mhi + i;
            ob_out[(size_t)o * NTOK + q] = acc[nt][i] + bb;   // [b][c][h*w]
        }
    }
}

// ---------------- host-side launch ----------------
extern "C" void kernel_launch(void* const* d_in, const int* in_sizes, int n_in,
                              void* d_out, int out_size, void* d_ws, size_t ws_size,
                              hipStream_t stream) {
    (void)in_sizes; (void)n_in; (void)out_size; (void)ws_size;
    const float* x  = (const float*)d_in[0];
    const float* w1 = (const float*)d_in[1];
    const float* b1 = (const float*)d_in[2];
    const float* w3 = (const float*)d_in[3];
    const float* b3 = (const float*)d_in[4];
    const float* w5 = (const float*)d_in[5];
    const float* b5 = (const float*)d_in[6];
    const float* qw = (const float*)d_in[7];
    const float* qb = (const float*)d_in[8];
    const float* kw = (const float*)d_in[9];
    const float* kb = (const float*)d_in[10];
    const float* vw = (const float*)d_in[11];
    const float* vb = (const float*)d_in[12];
    const float* ow = (const float*)d_in[13];
    const float* ob = (const float*)d_in[14];

    char* w = (char*)d_ws;
    size_t off = 0;
    auto take = [&](size_t bytes) -> char* {
        char* p = w + off;
        off = (off + bytes + 255) & ~(size_t)255;
        return p;
    };
    _Float16* fh    = (_Float16*)take((size_t)BATCH * NTOK * D3 * 2);
    _Float16* qwh   = (_Float16*)take(64 * D3 * 2);
    _Float16* kwh   = (_Float16*)take(64 * D3 * 2);
    _Float16* vwh   = (_Float16*)take(64 * D3 * 2);
    _Float16* owh   = (_Float16*)take(64 * 64 * 2);
    _Float16* Qh    = (_Float16*)take((size_t)BATCH * NTOK * 64 * 2);
    _Float16* Kh    = (_Float16*)take((size_t)BATCH * NTOK * 64 * 2);
    _Float16* Vh    = (_Float16*)take((size_t)BATCH * NTOK * 64 * 2);
    _Float16* S     = (_Float16*)take((size_t)NTOK * NTOK * 2);   // per-batch reuse
    float*    csum  = (float*)take(NTOK * 4);
    float*    mask  = (float*)take(NTOK * 4);
    _Float16* VmT   = (_Float16*)take((size_t)64 * NTOK * 2);
    _Float16* maskB = (_Float16*)take((size_t)16 * NTOK * 2);
    _Float16* yh    = (_Float16*)take((size_t)NTOK * 64 * 2);

    // weight prepack
    cvt_f16_kernel<<<(64 * D3 + 255) / 256, 256, 0, stream>>>(qw, qwh, 64 * D3);
    cvt_f16_kernel<<<(64 * D3 + 255) / 256, 256, 0, stream>>>(kw, kwh, 64 * D3);
    cvt_f16_kernel<<<(64 * D3 + 255) / 256, 256, 0, stream>>>(vw, vwh, 64 * D3);
    cvt_f16_kernel<<<(64 * 64 + 255) / 256, 256, 0, stream>>>(ow, owh, 64 * 64);

    // fused convolutions -> f_flat
    conv_kernel<<<(BATCH * CCH * NTOK) / 256, 256, 0, stream>>>(x, w1, b1, w3, b3, w5, b5, fh);

    // QKV projection for all batches (3 mats * 8 batches * 256 m-tiles waves)
    qkv_kernel<<<(3 * BATCH * 256) / 4, 128, 0, stream>>>(fh, qwh, kwh, vwh, qb, kb, vb, Qh, Kh, Vh);

    for (int bat = 0; bat < BATCH; ++bat) {
        logits_kernel<<<(256 * 32) / 4, 128, 0, stream>>>(Qh, Kh, S, bat);
        softmax_kernel<<<NTOK, 256, 0, stream>>>(S);
        colsum_kernel<<<NTOK / 256, 256, 0, stream>>>(S, csum);
        topk_kernel<<<1, 1024, 0, stream>>>(csum, mask);
        maskv_kernel<<<(64 * NTOK) / 256, 256, 0, stream>>>(Vh, mask, VmT, bat);
        maskb_kernel<<<(16 * NTOK) / 256, 256, 0, stream>>>(mask, maskB);
        av_kernel<<<256 / 4, 128, 0, stream>>>(S, VmT, maskB, yh);
        proj_kernel<<<256 / 4, 128, 0, stream>>>(yh, owh, ob, (float*)d_out, bat);
    }
}
